// hit_srf_86053964743292
// MI455X (gfx1250) — matrix operand order repaired and checked by reading, not compiled, not run on hardware
//
#include <hip/hip_runtime.h>

typedef _Float16 half_t;
typedef __attribute__((ext_vector_type(16))) _Float16 v16h;
typedef __attribute__((ext_vector_type(8)))  _Float16 h8;
typedef __attribute__((ext_vector_type(8)))  float    v8f;

// B=4, H=W=256, DIM=180, NH=6, CH=15 (padded to 16 -> 96 packed f16 channels),
// windows 16x16 -> NW=1024, L=256 tokens, BL=64 base tokens, C5=36 (padded 40).
#define NPIX   (4 * 256 * 256)
#define NW     1024
#define C5P    40      // 36 padded
#define C96    96      // 6 heads * 16 (15 padded)
#define CCAT   192     // 180 padded

union Frag { v16h v; h8 h[2]; half_t e[16]; };

struct HRow { const half_t* p; bool ok; };  // p is ALWAYS dereferenceable (>=8 halves)

__device__ __forceinline__ v8f wmma32(v16h a, v16h b, v8f c) {
  return __builtin_amdgcn_wmma_f32_16x16x32_f16(false, a, false, b, (short)0, c,
                                                false, false);
}

// ---- branchless fragment loaders; layouts per CDNA5 ISA 7.12.2 (wave32) ----
// A 16x32 f16: lane l: row m=l&15; K chunks {k0+kh..+7, k0+16+kh..+7}, kh=8*(l>>4)
// kvalid must be a multiple of 8 (all call sites satisfy this).
template <typename RowF>
__device__ __forceinline__ v16h load_a(RowF rowp, int k0, int kvalid) {
  const int lane = threadIdx.x & 31;
  const HRow r = rowp(lane & 15);
  const int kh = (lane >> 4) << 3;
  const h8 z = {};
  Frag f;
#pragma unroll
  for (int g = 0; g < 2; ++g) {
    const int kc = k0 + g * 16 + kh;
    const bool ok = r.ok && (kc + 8 <= kvalid);
    const h8 d = *(const h8*)(r.p + (ok ? kc : 0));  // unconditional, safe addr
    f.h[g] = ok ? d : z;                             // v_cndmask, no branch
  }
  return f.v;
}

// A fragment from f32 global rows (convert to f16): clamped-index, branchless.
template <typename RowF>
__device__ __forceinline__ v16h load_a_f32(RowF rowp, int k0, int kvalid) {
  const int lane = threadIdx.x & 31;
  const float* rp = rowp(lane & 15);
  const int kh = (lane >> 4) << 3;
  Frag f;
#pragma unroll
  for (int g = 0; g < 2; ++g)
#pragma unroll
    for (int j = 0; j < 8; ++j) {
      const int k = k0 + g * 16 + kh + j;
      const float v = rp[k < kvalid ? k : 0];        // unconditional load
      f.e[g * 8 + j] = (k < kvalid) ? (half_t)v : (half_t)0.0f;
    }
  return f.v;
}

// B 32x16 f16 (Bt row-major, K-contiguous): lane l: col n=l&15; K chunks {k0+kh..+7, k0+16+kh..+7}, kh=8*(l>>4)
template <typename RowF>
__device__ __forceinline__ v16h load_b(RowF rowp, int k0, int kvalid) {
  const int lane = threadIdx.x & 31;
  const HRow r = rowp(lane & 15);
  const int kh = (lane >> 4) << 3;
  const h8 z = {};
  Frag f;
#pragma unroll
  for (int g = 0; g < 2; ++g) {
    const int kc = k0 + g * 16 + kh;
    const bool ok = r.ok && (kc + 8 <= kvalid);
    const h8 d = *(const h8*)(r.p + (ok ? kc : 0));
    f.h[g] = ok ? d : z;
  }
  return f.v;
}

// Gather variants (strided K; channel-correlation transpose) — already branchless
template <typename EF>
__device__ __forceinline__ v16h load_a_gather(EF elem, int k0) {
  const int lane = threadIdx.x & 31;
  const int m = lane & 15, kh = (lane >> 4) << 3;
  Frag f;
#pragma unroll
  for (int g = 0; g < 2; ++g)
#pragma unroll
    for (int j = 0; j < 8; ++j) f.e[g * 8 + j] = elem(m, k0 + g * 16 + kh + j);
  return f.v;
}
template <typename EF>
__device__ __forceinline__ v16h load_b_gather(EF elem, int k0) {
  const int lane = threadIdx.x & 31;
  const int n = lane & 15, kh = (lane >> 4) << 3;
  Frag f;
#pragma unroll
  for (int j = 0; j < 16; ++j) f.e[j] = elem(n, k0 + (j >> 3) * 16 + kh + (j & 7));
  return f.v;
}

// ---------------- position-bias MLP + rpb ----------------
__device__ __forceinline__ void ln_relu11(const float* in, const float* g,
                                          const float* b, float* out) {
  float mu = 0.f;
  for (int j = 0; j < 11; ++j) mu += in[j];
  mu *= (1.f / 11.f);
  float var = 0.f;
  for (int j = 0; j < 11; ++j) { float d = in[j] - mu; var += d * d; }
  var *= (1.f / 11.f);
  const float inv = rsqrtf(var + 1e-5f);
  for (int j = 0; j < 11; ++j) {
    float y = (in[j] - mu) * inv * g[j] + b[j];
    out[j] = y > 0.f ? y : 0.f;
  }
}

__global__ __launch_bounds__(256) void pos_bias_k(
    const float* __restrict__ Wp, const float* __restrict__ bp,
    const float* __restrict__ g1, const float* __restrict__ b1,
    const float* __restrict__ W1, const float* __restrict__ c1,
    const float* __restrict__ g2, const float* __restrict__ b2,
    const float* __restrict__ W2, const float* __restrict__ c2,
    const float* __restrict__ g3, const float* __restrict__ b3,
    const float* __restrict__ W3, const float* __restrict__ c3,
    float* __restrict__ rpb /* [6][256][64] */) {
  __shared__ float pos_s[961 * 6];
  for (int row = threadIdx.x; row < 961; row += 256) {
    const float dy = (float)(row / 31 - 15);
    const float dx = (float)(row % 31 - 15);
    float p[11], q[11];
    for (int j = 0; j < 11; ++j) p[j] = dy * Wp[j * 2 + 0] + dx * Wp[j * 2 + 1] + bp[j];
    ln_relu11(p, g1, b1, q);
    for (int j = 0; j < 11; ++j) {
      float s = c1[j];
      for (int k = 0; k < 11; ++k) s += q[k] * W1[j * 11 + k];
      p[j] = s;
    }
    ln_relu11(p, g2, b2, q);
    for (int j = 0; j < 11; ++j) {
      float s = c2[j];
      for (int k = 0; k < 11; ++k) s += q[k] * W2[j * 11 + k];
      p[j] = s;
    }
    ln_relu11(p, g3, b3, q);
    for (int n = 0; n < 6; ++n) {
      float s = c3[n];
      for (int k = 0; k < 11; ++k) s += q[k] * W3[n * 11 + k];
      pos_s[row * 6 + n] = s;
    }
  }
  __syncthreads();
  for (int idx = threadIdx.x; idx < 6 * 256 * 64; idx += 256) {
    const int n = idx / (256 * 64);
    const int l = (idx >> 6) & 255;
    const int m = idx & 63;
    const int y1 = l >> 4, x1 = l & 15, by = m >> 3, bx = m & 7;
    float s = 0.f;
    for (int rh = 0; rh < 2; ++rh)
      for (int rw = 0; rw < 2; ++rw) {
        const int rpi = (y1 - (2 * by + rh) + 15) * 31 + (x1 - (2 * bx + rw) + 15);
        s += pos_s[rpi * 6 + n];
      }
    rpb[idx] = 0.25f * s;
  }
}

// ---------------- K1a: 1x1 conv 180->36 + leaky ----------------
__global__ __launch_bounds__(256) void conv1_k(const float* __restrict__ x,
                                               const float* __restrict__ w1,
                                               const float* __restrict__ b1,
                                               half_t* __restrict__ Y1) {
  __shared__ half_t Bt[48 * 192];  // Bt[o][k] = w1[k*36+o], zero padded
  for (int i = threadIdx.x; i < 48 * 192; i += 256) {
    const int o = i / 192, k = i % 192;
    Bt[i] = (o < 36 && k < 180) ? (half_t)w1[k * 36 + o] : (half_t)0.0f;
  }
  __syncthreads();
  const int wave = threadIdx.x >> 5, lane = threadIdx.x & 31;
  const long pix0 = (long)blockIdx.x * 256;
  for (int t = wave; t < 48; t += 8) {  // 16 Mtiles x 3 Ntiles
    const int mi = t / 3, ni = t % 3;
    v8f acc = {};
#pragma unroll
    for (int ks = 0; ks < 6; ++ks) {
      v16h a = load_a_f32([&](int m) { return x + (pix0 + mi * 16 + m) * 180; },
                          ks * 32, 180);
      v16h bb = load_b([&](int n) { return HRow{Bt + (ni * 16 + n) * 192, true}; },
                       ks * 32, 192);
      acc = wmma32(a, bb, acc);
    }
    const int o = ni * 16 + (lane & 15);
    for (int r = 0; r < 8; ++r) {
      const long pix = pix0 + mi * 16 + r + ((lane >> 4) << 3);
      if (o < 36) {
        float v = acc[r] + b1[o];
        v = v > 0.f ? v : 0.2f * v;
        Y1[pix * C5P + o] = (half_t)v;
      } else if (o < 40) {
        Y1[pix * C5P + o] = (half_t)0.0f;  // zero pad slots 36..39
      }
    }
  }
}

// ---------------- K1b: 3x3 conv 36->36 + leaky (9 shifted GEMMs) ----------------
__global__ __launch_bounds__(256) void conv2_k(const half_t* __restrict__ Y1,
                                               const float* __restrict__ w2,
                                               const float* __restrict__ b2,
                                               half_t* __restrict__ Y2) {
  __shared__ half_t Bt[9 * 48 * C5P];  // Bt[tap][o][i] = w2[tap][i][o]
  for (int i = threadIdx.x; i < 9 * 48 * C5P; i += 256) {
    const int tap = i / (48 * C5P);
    const int o = (i / C5P) % 48;
    const int k = i % C5P;
    Bt[i] = (o < 36 && k < 36) ? (half_t)w2[(tap * 36 + k) * 36 + o] : (half_t)0.0f;
  }
  __syncthreads();
  const int bb = blockIdx.x >> 8, y0 = ((blockIdx.x >> 4) & 15) * 16,
            x0 = (blockIdx.x & 15) * 16;
  const int wave = threadIdx.x >> 5, lane = threadIdx.x & 31;
  for (int t = wave; t < 48; t += 8) {
    const int mi = t / 3, ni = t % 3;
    v8f acc = {};
    for (int tap = 0; tap < 9; ++tap) {
      const int dy = tap / 3 - 1, dx = tap % 3 - 1;
#pragma unroll
      for (int ks = 0; ks < 2; ++ks) {
        v16h a = load_a(
            [&](int m) -> HRow {
              const int p = mi * 16 + m;
              int gy = y0 + (p >> 4) + dy, gx = x0 + (p & 15) + dx;
              const bool ok = ((unsigned)gy < 256u) && ((unsigned)gx < 256u);
              gy = gy < 0 ? 0 : (gy > 255 ? 255 : gy);
              gx = gx < 0 ? 0 : (gx > 255 ? 255 : gx);
              return HRow{Y1 + (((long)bb * 256 + gy) * 256 + gx) * C5P, ok};
            },
            ks * 32, C5P);
        v16h bf = load_b(
            [&](int n) { return HRow{Bt + (tap * 48 + ni * 16 + n) * C5P, true}; },
            ks * 32, C5P);
        acc = wmma32(a, bf, acc);
      }
    }
    const int o = ni * 16 + (lane & 15);
    for (int r = 0; r < 8; ++r) {
      const int p = mi * 16 + r + ((lane >> 4) << 3);
      const long pix = ((long)bb * 256 + y0 + (p >> 4)) * 256 + x0 + (p & 15);
      if (o < 36) {
        float v = acc[r] + b2[o];
        v = v > 0.f ? v : 0.2f * v;
        Y2[pix * C5P + o] = (half_t)v;
      } else if (o < 40) {
        Y2[pix * C5P + o] = (half_t)0.0f;
      }
    }
  }
}

// ------- K1c: lin = x@Wlin (staged in 64-row chunks), t3 = Y2@W3, qv = t3*lin -------
__global__ __launch_bounds__(256) void dfe_out_k(
    const float* __restrict__ x, const half_t* __restrict__ Y2,
    const float* __restrict__ w3, const float* __restrict__ b3,
    const float* __restrict__ wl, const float* __restrict__ blin,
    half_t* __restrict__ qg, half_t* __restrict__ vg,
    half_t* __restrict__ linbuf /* xcat region */) {
  __shared__ half_t WS[64 * 192];
  const int win = blockIdx.x;
  const int bb = win >> 8, wy = (win >> 4) & 15, wx = win & 15;
  const int wave = threadIdx.x >> 5, lane = threadIdx.x & 31;
  half_t* lb = linbuf + (long)win * 256 * CCAT;
  auto xrow = [&](int tok) {
    return x + (((long)bb * 256 + wy * 16 + (tok >> 4)) * 256 + wx * 16 + (tok & 15)) * 180;
  };
  // phase 1: linear path, 3 output groups of 64
  for (int og = 0; og < 3; ++og) {
    __syncthreads();
    for (int i = threadIdx.x; i < 64 * 192; i += 256) {
      const int o = og * 64 + i / 192, k = i % 192;
      WS[i] = (o < 180 && k < 180) ? (half_t)wl[k * 180 + o] : (half_t)0.0f;
    }
    __syncthreads();
    for (int t = wave; t < 64; t += 8) {  // 16 Mtiles x 4 Ntiles
      const int mi = t >> 2, ni = t & 3;
      v8f acc = {};
#pragma unroll
      for (int ks = 0; ks < 6; ++ks) {
        v16h a = load_a_f32([&](int m) { return xrow(mi * 16 + m); }, ks * 32, 180);
        v16h bf = load_b([&](int n) { return HRow{WS + (ni * 16 + n) * 192, true}; },
                         ks * 32, 192);
        acc = wmma32(a, bf, acc);
      }
      const int o = og * 64 + ni * 16 + (lane & 15);
      for (int r = 0; r < 8; ++r) {
        const int tok = mi * 16 + r + ((lane >> 4) << 3);
        half_t v = (half_t)0.0f;
        if (o < 180) v = (half_t)(acc[r] + blin[o]);
        lb[tok * CCAT + o] = v;  // o in 180..191 get zeros
      }
    }
  }
  __syncthreads();
  __threadfence();
  __syncthreads();
  // phase 2: 1x1 conv 36->180, gate with lin, scatter to qg/vg
  for (int i = threadIdx.x; i < 192 * C5P; i += 256) {
    const int o = i / C5P, k = i % C5P;
    WS[i] = (o < 180 && k < 36) ? (half_t)w3[k * 180 + o] : (half_t)0.0f;
  }
  __syncthreads();
  for (int t = wave; t < 192; t += 8) {  // 16 Mtiles x 12 Ntiles
    const int mi = t / 12, ni = t % 12;
    v8f acc = {};
#pragma unroll
    for (int ks = 0; ks < 2; ++ks) {
      v16h a = load_a(
          [&](int m) -> HRow {
            const int tok = mi * 16 + m;
            return HRow{Y2 + (((long)bb * 256 + wy * 16 + (tok >> 4)) * 256 + wx * 16 +
                              (tok & 15)) * C5P, true};
          },
          ks * 32, C5P);
      v16h bf = load_b([&](int n) { return HRow{WS + (ni * 16 + n) * C5P, true}; },
                       ks * 32, C5P);
      acc = wmma32(a, bf, acc);
    }
    const int o = ni * 16 + (lane & 15);
    if (o < 180) {
      const int s = o / 90, cc = o % 90, nh = cc / 15, ch = cc % 15;
      half_t* dst = s ? vg : qg;
      for (int r = 0; r < 8; ++r) {
        const int tok = mi * 16 + r + ((lane >> 4) << 3);
        const float qv = (acc[r] + b3[o]) * (float)lb[tok * CCAT + o];
        dst[((long)win * 256 + tok) * C96 + nh * 16 + ch] = (half_t)qv;
      }
    }
  }
  // zero the per-head pad slot (ch == 15)
  for (int i = threadIdx.x; i < 256 * 12; i += 256) {
    const int tok = i / 12, rest = i % 12, s = rest / 6, nh = rest % 6;
    (s ? vg : qg)[((long)win * 256 + tok) * C96 + nh * 16 + 15] = (half_t)0.0f;
  }
}

// ---------------- K2: spatial self-correlation per window ----------------
__global__ __launch_bounds__(256) void spatial_k(
    const half_t* __restrict__ qg, const half_t* __restrict__ vg,
    const float* __restrict__ rpb, const float* __restrict__ slw,
    const float* __restrict__ slb, half_t* __restrict__ xcat) {
  __shared__ half_t vp[6 * 64 * 16];   // [head][m][ch]
  __shared__ half_t vpT[6 * 16 * 64];  // [head][ch][m]
  __shared__ half_t corr[256 * 64];    // per-head, reused
  const int win = blockIdx.x;
  const half_t* q = qg + (long)win * 256 * C96;
  const half_t* v = vg + (long)win * 256 * C96;
  const float w0 = slw[0], w1 = slw[1], w2 = slw[2], w3 = slw[3], sb = slb[0];
  for (int i = threadIdx.x; i < 6 * 64 * 16; i += 256) {
    const int n = i / (64 * 16), m = (i / 16) & 63, ch = i & 15;
    half_t val = (half_t)0.0f;
    if (ch < 15) {
      const int r0 = ((m >> 3) * 2) * 16 + (m & 7) * 2;
      const int c = n * 16 + ch;
      const float s = w0 * (float)v[r0 * C96 + c] + w1 * (float)v[(r0 + 1) * C96 + c] +
                      w2 * (float)v[(r0 + 16) * C96 + c] +
                      w3 * (float)v[(r0 + 17) * C96 + c] + sb;
      val = (half_t)s;
    }
    vp[i] = val;
    vpT[(n * 16 + ch) * 64 + m] = val;
  }
  for (int i = threadIdx.x; i < 256 * 12; i += 256)
    xcat[((long)win * 256 + i / 12) * CCAT + 180 + i % 12] = (half_t)0.0f;
  __syncthreads();
  const int wave = threadIdx.x >> 5, lane = threadIdx.x & 31;
  for (int n = 0; n < 6; ++n) {
    for (int t = wave; t < 64; t += 8) {  // corr: 16 Mtiles x 4 Ntiles, K=16(pad)
      const int mi = t >> 2, ni = t & 3;
      v8f acc = {};
      v16h a = load_a(
          [&](int m) { return HRow{q + (mi * 16 + m) * C96 + n * 16, true}; }, 0, 16);
      v16h bf = load_b(
          [&](int nn) { return HRow{vp + (n * 64 + ni * 16 + nn) * 16, true}; }, 0, 16);
      acc = wmma32(a, bf, acc);
      const int col = ni * 16 + (lane & 15);
      for (int r = 0; r < 8; ++r) {
        const int row = mi * 16 + r + ((lane >> 4) << 3);
        corr[row * 64 + col] =
            (half_t)(acc[r] * (1.f / 15.f) + rpb[(n * 256 + row) * 64 + col]);
      }
    }
    __syncthreads();
    for (int mi = wave; mi < 16; mi += 8) {  // x_sp: 16 Mtiles x 1 Ntile, K=64
      v8f acc = {};
#pragma unroll
      for (int ks = 0; ks < 2; ++ks) {
        v16h a = load_a([&](int m) { return HRow{corr + (mi * 16 + m) * 64, true}; },
                        ks * 32, 64);
        v16h bf = load_b([&](int nn) { return HRow{vpT + (n * 16 + nn) * 64, true}; },
                         ks * 32, 64);
        acc = wmma32(a, bf, acc);
      }
      const int ch = lane & 15;
      if (ch < 15) {
        for (int r = 0; r < 8; ++r) {
          const int row = mi * 16 + r + ((lane >> 4) << 3);
          xcat[((long)win * 256 + row) * CCAT + n * 15 + ch] = (half_t)acc[r];
        }
      }
    }
    __syncthreads();
  }
}

// ---------------- K3: channel self-correlation per window ----------------
__global__ __launch_bounds__(256) void channel_k(const half_t* __restrict__ qg,
                                                 const half_t* __restrict__ vg,
                                                 half_t* __restrict__ xcat) {
  __shared__ half_t cmap[96 * 96];  // Bt[c'][d'] f16
  const int win = blockIdx.x;
  const half_t* q = qg + (long)win * 256 * C96;
  const half_t* v = vg + (long)win * 256 * C96;
  const int wave = threadIdx.x >> 5, lane = threadIdx.x & 31;
  for (int t = wave; t < 36; t += 8) {  // cmap: 6x6 tiles, K=256 (gather transpose)
    const int mi = t / 6, ni = t % 6;
    v8f acc = {};
    for (int ks = 0; ks < 8; ++ks) {
      v16h a = load_a_gather(
          [&](int m, int k) { return q[(long)k * C96 + mi * 16 + m]; }, ks * 32);
      v16h bf = load_b_gather(
          [&](int n, int k) { return v[(long)k * C96 + ni * 16 + n]; }, ks * 32);
      acc = wmma32(a, bf, acc);
    }
    const int col = ni * 16 + (lane & 15);
    for (int r = 0; r < 8; ++r) {
      const int row = mi * 16 + r + ((lane >> 4) << 3);
      cmap[row * 96 + col] = (half_t)(acc[r] * (1.f / 256.f));
    }
  }
  __syncthreads();
  for (int t = wave; t < 96; t += 8) {  // x_ch: 16 Mtiles x 6 Ntiles, K=96
    const int mi = t / 6, ni = t % 6;
    v8f acc = {};
#pragma unroll
    for (int ks = 0; ks < 3; ++ks) {
      v16h a = load_a([&](int m) { return HRow{v + (mi * 16 + m) * C96, true}; },
                      ks * 32, 96);
      v16h bf = load_b([&](int n) { return HRow{cmap + (ni * 16 + n) * 96, true}; },
                       ks * 32, 96);
      acc = wmma32(a, bf, acc);
    }
    const int cp = ni * 16 + (lane & 15);
    const int nh = cp >> 4, ch = cp & 15;
    if (ch < 15) {
      for (int r = 0; r < 8; ++r) {
        const int row = mi * 16 + r + ((lane >> 4) << 3);
        xcat[((long)win * 256 + row) * CCAT + 90 + nh * 15 + ch] = (half_t)acc[r];
      }
    }
  }
}

// ---------------- K4: final 180x180 projection + window reverse ----------------
__global__ __launch_bounds__(256) void proj_k(const half_t* __restrict__ xcat,
                                              const float* __restrict__ pw,
                                              const float* __restrict__ pb,
                                              float* __restrict__ out) {
  __shared__ half_t PW[64 * 192];
  const int win = blockIdx.x;
  const int bb = win >> 8, wy = (win >> 4) & 15, wx = win & 15;
  const half_t* xr = xcat + (long)win * 256 * CCAT;
  const int wave = threadIdx.x >> 5, lane = threadIdx.x & 31;
  for (int og = 0; og < 3; ++og) {
    __syncthreads();
    for (int i = threadIdx.x; i < 64 * 192; i += 256) {
      const int o = og * 64 + i / 192, k = i % 192;
      PW[i] = (o < 180 && k < 180) ? (half_t)pw[o * 180 + k] : (half_t)0.0f;
    }
    __syncthreads();
    for (int t = wave; t < 64; t += 8) {  // 16 Mtiles x 4 Ntiles, K=192
      const int mi = t >> 2, ni = t & 3;
      v8f acc = {};
#pragma unroll
      for (int ks = 0; ks < 6; ++ks) {
        v16h a = load_a([&](int m) { return HRow{xr + (mi * 16 + m) * CCAT, true}; },
                        ks * 32, 192);
        v16h bf = load_b([&](int n) { return HRow{PW + (ni * 16 + n) * 192, true}; },
                         ks * 32, 192);
        acc = wmma32(a, bf, acc);
      }
      const int o = og * 64 + ni * 16 + (lane & 15);
      if (o < 180) {
        const float bias = pb[o];
        for (int r = 0; r < 8; ++r) {
          const int tok = mi * 16 + r + ((lane >> 4) << 3);
          const int y = wy * 16 + (tok >> 4), xx = wx * 16 + (tok & 15);
          out[(((long)bb * 256 + y) * 256 + xx) * 180 + o] = acc[r] + bias;
        }
      }
    }
  }
}

// ---------------- launch ----------------
extern "C" void kernel_launch(void* const* d_in, const int* in_sizes, int n_in,
                              void* d_out, int out_size, void* d_ws, size_t ws_size,
                              hipStream_t stream) {
  (void)in_sizes; (void)n_in; (void)out_size; (void)ws_size;
  const float* x    = (const float*)d_in[0];
  const float* c1w  = (const float*)d_in[1];
  const float* c1b  = (const float*)d_in[2];
  const float* c2w  = (const float*)d_in[3];
  const float* c2b  = (const float*)d_in[4];
  const float* c3w  = (const float*)d_in[5];
  const float* c3b  = (const float*)d_in[6];
  const float* linw = (const float*)d_in[7];
  const float* linb = (const float*)d_in[8];
  const float* slw  = (const float*)d_in[9];
  const float* slb  = (const float*)d_in[10];
  const float* ppw  = (const float*)d_in[11];
  const float* ppb  = (const float*)d_in[12];
  const float* g1   = (const float*)d_in[13];
  const float* b1   = (const float*)d_in[14];
  const float* W1   = (const float*)d_in[15];
  const float* cb1  = (const float*)d_in[16];
  const float* g2   = (const float*)d_in[17];
  const float* b2   = (const float*)d_in[18];
  const float* W2   = (const float*)d_in[19];
  const float* cb2  = (const float*)d_in[20];
  const float* g3   = (const float*)d_in[21];
  const float* b3   = (const float*)d_in[22];
  const float* W3   = (const float*)d_in[23];
  const float* cb3  = (const float*)d_in[24];
  const float* pjw  = (const float*)d_in[25];
  const float* pjb  = (const float*)d_in[26];
  float* out = (float*)d_out;

  char* ws = (char*)d_ws;
  const size_t szY  = (size_t)NPIX * C5P * 2;       // 20.97 MB each
  const size_t szQV = (size_t)NW * 256 * C96 * 2;   // 50.3 MB each
  const size_t szXC = (size_t)NW * 256 * CCAT * 2;  // 100.7 MB
  half_t* Y1   = (half_t*)(ws);
  half_t* Y2   = (half_t*)(ws + szY);
  half_t* qg   = (half_t*)(ws + 2 * szY);
  half_t* vg   = (half_t*)(ws + 2 * szY + szQV);
  half_t* xcat = (half_t*)(ws + 2 * szY + 2 * szQV);
  float*  rpb  = (float*)(ws + 2 * szY + 2 * szQV + szXC);

  pos_bias_k<<<1, 256, 0, stream>>>(ppw, ppb, g1, b1, W1, cb1, g2, b2, W2, cb2,
                                    g3, b3, W3, cb3, rpb);
  conv1_k<<<NPIX / 256, 256, 0, stream>>>(x, c1w, c1b, Y1);
  conv2_k<<<NW, 256, 0, stream>>>(Y1, c2w, c2b, Y2);
  dfe_out_k<<<NW, 256, 0, stream>>>(x, Y2, c3w, c3b, linw, linb, qg, vg, xcat);
  spatial_k<<<NW, 256, 0, stream>>>(qg, vg, rpb, slw, slb, xcat);
  channel_k<<<NW, 256, 0, stream>>>(qg, vg, xcat);
  proj_k<<<NW, 256, 0, stream>>>(xcat, pjw, pjb, out);
}
